// DecoderRNN_35055523070181
// MI455X (gfx1250) — matrix operand order repaired and checked by reading, not compiled, not run on hardware
//
#include <hip/hip_runtime.h>
#include <hip/hip_bf16.h>
#include <math.h>

// ---------- problem constants ----------
#define BB   128          // batch
#define TT   10           // MAX_LENGTH
#define HH   512          // hidden
#define VV   50257        // vocab
#define G3H  1536         // 3*H
#define MM   (BB*TT)      // 1280 GEMM rows (r = b*T + t)

// ---------- CDNA5 WMMA types ----------
typedef __attribute__((ext_vector_type(16))) __bf16 v16bf;
typedef __attribute__((ext_vector_type(8)))  float  v8f;

struct __attribute__((aligned(16))) U4 { unsigned int x, y, z, w; };
union Frag { v16bf v; U4 q[2]; };

__device__ __forceinline__ unsigned short f32_to_bf16_rne(float f) {
  union { float f; unsigned int u; } c; c.f = f;
  unsigned int r = c.u + 0x7FFFu + ((c.u >> 16) & 1u);
  return (unsigned short)(r >> 16);
}
__device__ __forceinline__ float sigmoidf_(float x) { return 1.0f / (1.0f + expf(-x)); }

__device__ __forceinline__ v8f v8f_zero() {
  v8f z;
  #pragma unroll
  for (int e = 0; e < 8; ++e) z[e] = 0.0f;
  return z;
}

// ---------- CDNA5 async global->LDS (ASYNCcnt path, no VGPR round-trip) ----
__device__ __forceinline__ void async_ld16(unsigned lds_off, unsigned long long gaddr) {
  // GLOBAL_LOAD_ASYNC_TO_LDS_B128: VDST = per-lane LDS byte address,
  // VADDR = per-lane 64-bit global address (GV mode).
  asm volatile("global_load_async_to_lds_b128 %0, %1, off"
               :: "v"(lds_off), "v"(gaddr) : "memory");
}
__device__ __forceinline__ void async_wait0() {
  asm volatile("s_wait_asynccnt 0x0" ::: "memory");
}
__device__ __forceinline__ unsigned lds_off32(const void* p) {
  // flat pointer in LDS aperture: addr[31:0] is the LDS byte offset
  return (unsigned)(unsigned long long)p;
}

// =====================================================================
// Generic NT GEMM: C[M,N] = A[M,K](bf16) * W[N,K](bf16)^T + bias[N]
// Block tile 128x128, K-chunk 32, 256 threads (8 waves), wave tile 32x64.
// Double-buffered LDS fed by async loads; one barrier per K-chunk.
// =====================================================================
__global__ __launch_bounds__(256) void gemm_nt_bf16(
    const unsigned short* __restrict__ A,   // [M,K] bf16 bits
    const unsigned short* __restrict__ W,   // [N,K] bf16 bits (row-clamped)
    const float*          __restrict__ bias,// [N]
    float*                __restrict__ C,   // [M,N]
    int M, int N, int K)
{
  __shared__ __align__(16) unsigned short lA[2][128][32];
  __shared__ __align__(16) unsigned short lB[2][128][32];

  const int tid  = threadIdx.x;
  const int lane = tid & 31;
  const int wave = tid >> 5;
  const int wm   = wave & 3;   // M sub-tile: 32 rows each
  const int wn   = wave >> 2;  // N sub-tile: 64 cols each
  const int half = lane >> 4;
  const int lan  = lane & 15;
  const int m0   = blockIdx.y * 128;
  const int n0   = blockIdx.x * 128;

  // this thread's two staging chunks (16B each per tile)
  const int c0 = tid, c1 = tid + 256;
  const int ra0 = c0 >> 2, oa0 = (c0 & 3) * 8;
  const int ra1 = c1 >> 2, oa1 = (c1 & 3) * 8;
  int gn0 = n0 + ra0; if (gn0 > N - 1) gn0 = N - 1;  // clamp vocab tail:
  int gn1 = n0 + ra1; if (gn1 > N - 1) gn1 = N - 1;  // OOB cols never stored

  v8f acc[2][4];
  #pragma unroll
  for (int i = 0; i < 2; ++i)
    #pragma unroll
    for (int j = 0; j < 4; ++j) acc[i][j] = v8f_zero();

  auto stage = [&](int buf, int k0) {
    async_ld16(lds_off32(&lA[buf][ra0][oa0]),
               (unsigned long long)(const void*)&A[(size_t)(m0 + ra0) * K + k0 + oa0]);
    async_ld16(lds_off32(&lA[buf][ra1][oa1]),
               (unsigned long long)(const void*)&A[(size_t)(m0 + ra1) * K + k0 + oa1]);
    async_ld16(lds_off32(&lB[buf][ra0][oa0]),
               (unsigned long long)(const void*)&W[(size_t)gn0 * K + k0 + oa0]);
    async_ld16(lds_off32(&lB[buf][ra1][oa1]),
               (unsigned long long)(const void*)&W[(size_t)gn1 * K + k0 + oa1]);
  };

  stage(0, 0);
  int cur = 0;
  for (int k0 = 0; k0 < K; k0 += 32) {
    async_wait0();        // my async writes for chunk k are in LDS
    __syncthreads();      // everyone's are
    if (k0 + 32 < K) stage(cur ^ 1, k0 + 32);   // prefetch next chunk

    // ---- fragments per ISA VGPR layouts ----
    Frag a[2], b[4];
    #pragma unroll
    for (int i = 0; i < 2; ++i) {
      int m = wm * 32 + i * 16 + lan;                 // lane&15 = M row
      a[i].q[0] = *(const U4*)&lA[cur][m][half * 8];  // K 0..7 / 8..15
      a[i].q[1] = *(const U4*)&lA[cur][m][16 + half * 8];
    }
    #pragma unroll
    for (int j = 0; j < 4; ++j) {
      int n = wn * 64 + j * 16 + lan;                 // lane&15 = N col
      b[j].q[0] = *(const U4*)&lB[cur][n][half * 8];  // K 0..7 / 8..15
      b[j].q[1] = *(const U4*)&lB[cur][n][16 + half * 8];
    }
    #pragma unroll
    for (int i = 0; i < 2; ++i)
      #pragma unroll
      for (int j = 0; j < 4; ++j)
        acc[i][j] = __builtin_amdgcn_wmma_f32_16x16x32_bf16(
            false, a[i].v, false, b[j].v, (short)0, acc[i][j], false, false);
    cur ^= 1;
  }

  // ---- epilogue: C/D layout M = half*8 + vgpr, N = lane&15 ----
  #pragma unroll
  for (int i = 0; i < 2; ++i)
    #pragma unroll
    for (int j = 0; j < 4; ++j) {
      int n = n0 + wn * 64 + j * 16 + lan;
      if (n < N) {
        float bv = bias[n];
        #pragma unroll
        for (int e = 0; e < 8; ++e) {
          int m = m0 + wm * 32 + i * 16 + half * 8 + e;
          C[(size_t)m * N + n] = acc[i][j][e] + bv;
        }
      }
    }
}

// =====================================================================
// GRU step: gh = h @ w_hh^T (+b_hh), fused gate combine.
// Block handles 64 columns of H for all 3 gates. Grid = H/64 = 8 blocks.
// =====================================================================
__global__ __launch_bounds__(256) void gru_step(
    const unsigned short* __restrict__ hb,    // prev h bf16 [128,512]
    const float*          __restrict__ hf,    // prev h f32  [128,512]
    const unsigned short* __restrict__ whh,   // bf16 [1536,512]
    const float*          __restrict__ bhh,   // [1536]
    const float*          __restrict__ gx,    // [1280,1536] (b_ih included)
    float*                __restrict__ hf_out,
    unsigned short*       __restrict__ hb_out,
    unsigned short*       __restrict__ outs,  // bf16 [1280,512]
    int t)
{
  __shared__ __align__(16) unsigned short lA[2][128][32];
  __shared__ __align__(16) unsigned short lB[2][192][32];  // 3 gates x 64 cols

  const int tid  = threadIdx.x;
  const int lane = tid & 31;
  const int wave = tid >> 5;
  const int wm   = wave & 3;   // 32 batch rows each
  const int wn   = wave >> 2;  // 32 cols each (block covers 64)
  const int half = lane >> 4;
  const int lan  = lane & 15;
  const int j0   = blockIdx.x * 64;

  // staging assignments: A = 512 chunks (2/thread), B = 768 chunks (3/thread)
  const int ca0 = tid, ca1 = tid + 256;
  const int ra0 = ca0 >> 2, oa0 = (ca0 & 3) * 8;
  const int ra1 = ca1 >> 2, oa1 = (ca1 & 3) * 8;
  int rb[3], ob[3], gb[3];
  #pragma unroll
  for (int q = 0; q < 3; ++q) {
    int c = tid + q * 256;
    rb[q] = c >> 2; ob[q] = (c & 3) * 8;
    int g = rb[q] >> 6, cc = rb[q] & 63;
    gb[q] = g * HH + j0 + cc;                 // row in whh
  }

  v8f acc[3][2][2];
  #pragma unroll
  for (int g = 0; g < 3; ++g)
    #pragma unroll
    for (int i = 0; i < 2; ++i)
      #pragma unroll
      for (int j = 0; j < 2; ++j) acc[g][i][j] = v8f_zero();

  auto stage = [&](int buf, int k0) {
    async_ld16(lds_off32(&lA[buf][ra0][oa0]),
               (unsigned long long)(const void*)&hb[(size_t)ra0 * HH + k0 + oa0]);
    async_ld16(lds_off32(&lA[buf][ra1][oa1]),
               (unsigned long long)(const void*)&hb[(size_t)ra1 * HH + k0 + oa1]);
    #pragma unroll
    for (int q = 0; q < 3; ++q)
      async_ld16(lds_off32(&lB[buf][rb[q]][ob[q]]),
                 (unsigned long long)(const void*)&whh[(size_t)gb[q] * HH + k0 + ob[q]]);
  };

  stage(0, 0);
  int cur = 0;
  for (int k0 = 0; k0 < HH; k0 += 32) {
    async_wait0();
    __syncthreads();
    if (k0 + 32 < HH) stage(cur ^ 1, k0 + 32);

    Frag a[2], b[3][2];
    #pragma unroll
    for (int i = 0; i < 2; ++i) {
      int m = wm * 32 + i * 16 + lan;
      a[i].q[0] = *(const U4*)&lA[cur][m][half * 8];
      a[i].q[1] = *(const U4*)&lA[cur][m][16 + half * 8];
    }
    #pragma unroll
    for (int g = 0; g < 3; ++g)
      #pragma unroll
      for (int j = 0; j < 2; ++j) {
        int n = g * 64 + wn * 32 + j * 16 + lan;
        b[g][j].q[0] = *(const U4*)&lB[cur][n][half * 8];
        b[g][j].q[1] = *(const U4*)&lB[cur][n][16 + half * 8];
      }
    #pragma unroll
    for (int g = 0; g < 3; ++g)
      #pragma unroll
      for (int i = 0; i < 2; ++i)
        #pragma unroll
        for (int j = 0; j < 2; ++j)
          acc[g][i][j] = __builtin_amdgcn_wmma_f32_16x16x32_bf16(
              false, a[i].v, false, b[g][j].v, (short)0, acc[g][i][j], false, false);
    cur ^= 1;
  }

  // ---- fused gate combine + h update ----
  #pragma unroll
  for (int i = 0; i < 2; ++i)
    #pragma unroll
    for (int j = 0; j < 2; ++j) {
      int jcol = j0 + wn * 32 + j * 16 + lan;
      float br = bhh[jcol], bz = bhh[HH + jcol], bn = bhh[2 * HH + jcol];
      #pragma unroll
      for (int e = 0; e < 8; ++e) {
        int b_ = wm * 32 + i * 16 + half * 8 + e;      // batch row
        size_t gi = ((size_t)b_ * TT + t) * G3H;
        float ghr = acc[0][i][j][e] + br;
        float ghz = acc[1][i][j][e] + bz;
        float ghn = acc[2][i][j][e] + bn;
        float r = sigmoidf_(gx[gi + jcol] + ghr);
        float z = sigmoidf_(gx[gi + HH + jcol] + ghz);
        float n = tanhf(gx[gi + 2 * HH + jcol] + r * ghn);
        float hp = hf[(size_t)b_ * HH + jcol];
        float hn = (1.0f - z) * n + z * hp;
        hf_out[(size_t)b_ * HH + jcol] = hn;
        unsigned short hb16 = f32_to_bf16_rne(hn);
        hb_out[(size_t)b_ * HH + jcol] = hb16;
        outs[((size_t)b_ * TT + t) * HH + jcol] = hb16;
      }
    }
}

// =====================================================================
// Small helper kernels
// =====================================================================
__global__ void cvt_f32_bf16(const float* __restrict__ src,
                             unsigned short* __restrict__ dst, int n) {
  int i = blockIdx.x * 256 + threadIdx.x;
  if (i < n) dst[i] = f32_to_bf16_rne(src[i]);
}

__global__ void embed_relu_bf16(const int* __restrict__ target,
                                const float* __restrict__ emb,
                                unsigned short* __restrict__ xb) {
  int idx = blockIdx.x * 256 + threadIdx.x;       // [1280*512]
  if (idx >= MM * HH) return;
  int k = idx & (HH - 1);
  int r = idx >> 9;                                // r = b*T + t
  int t = r % TT, b = r / TT;
  int tok = (t == 0) ? 0 : target[b * TT + t - 1]; // teacher forcing, BOS=0
  float x = emb[(size_t)tok * HH + k];
  x = x > 0.0f ? x : 0.0f;
  xb[idx] = f32_to_bf16_rne(x);
}

__global__ void init_h(const float* __restrict__ hidden,
                       float* __restrict__ hf, unsigned short* __restrict__ hb) {
  int i = blockIdx.x * 256 + threadIdx.x;
  if (i < BB * HH) { float v = hidden[i]; hf[i] = v; hb[i] = f32_to_bf16_rne(v); }
}

// one block per row: online (max, sum-exp) reduction -> lse
__global__ __launch_bounds__(256) void row_lse(const float* __restrict__ C,
                                               float* __restrict__ lse, int N) {
  const float* row = C + (size_t)blockIdx.x * N;
  int tid = threadIdx.x;
  float m = -INFINITY, s = 0.0f;
  for (int v = tid; v < N; v += 256) {
    float x = row[v];
    if (x > m) { s = s * expf(m - x) + 1.0f; m = x; }
    else       { s += expf(x - m); }
  }
  __shared__ float sm[256], ss[256];
  sm[tid] = m; ss[tid] = s; __syncthreads();
  for (int ofs = 128; ofs > 0; ofs >>= 1) {
    if (tid < ofs) {
      float m2 = sm[tid + ofs], s2 = ss[tid + ofs];
      float mm = fmaxf(sm[tid], m2);
      ss[tid] = ss[tid] * expf(sm[tid] - mm) + s2 * expf(m2 - mm);
      sm[tid] = mm;
    }
    __syncthreads();
  }
  if (tid == 0) lse[blockIdx.x] = sm[0] + logf(ss[0]);
}

__global__ void sub_lse(float* __restrict__ C, const float* __restrict__ lse, int N) {
  int col = blockIdx.x * 256 + threadIdx.x;
  int row = blockIdx.y;
  if (col < N) C[(size_t)row * N + col] -= lse[row];
}

__global__ void copy_hlast(const float* __restrict__ hf, float* __restrict__ out) {
  int i = blockIdx.x * 256 + threadIdx.x;
  if (i < BB * HH) out[i] = hf[i];
}

// =====================================================================
// Host launch
// =====================================================================
extern "C" void kernel_launch(void* const* d_in, const int* in_sizes, int n_in,
                              void* d_out, int out_size, void* d_ws, size_t ws_size,
                              hipStream_t stream) {
  const float* hidden = (const float*)d_in[1];
  const int*   target = (const int*)  d_in[2];
  const float* emb    = (const float*)d_in[3];
  const float* w_ih   = (const float*)d_in[4];
  const float* w_hh   = (const float*)d_in[5];
  const float* b_ih   = (const float*)d_in[6];
  const float* b_hh   = (const float*)d_in[7];
  const float* w_out  = (const float*)d_in[8];
  const float* b_out  = (const float*)d_in[9];
  float* out = (float*)d_out;

  // workspace carve-up (256B aligned)
  char* ws = (char*)d_ws;
  size_t off = 0;
  auto carve = [&](size_t bytes) {
    size_t o = off; off = (off + bytes + 255) & ~(size_t)255; return o;
  };
  size_t o_xb    = carve((size_t)MM * HH * 2);        // embed bf16
  size_t o_outs  = carve((size_t)MM * HH * 2);        // GRU outs bf16
  size_t o_wih   = carve((size_t)G3H * HH * 2);
  size_t o_whh   = carve((size_t)G3H * HH * 2);
  size_t o_wout  = carve((size_t)VV * HH * 2);
  size_t o_gx    = carve((size_t)MM * G3H * 4);
  size_t o_hf0   = carve((size_t)BB * HH * 4);
  size_t o_hf1   = carve((size_t)BB * HH * 4);
  size_t o_hb0   = carve((size_t)BB * HH * 2);
  size_t o_hb1   = carve((size_t)BB * HH * 2);
  size_t o_lse   = carve((size_t)MM * 4);
  if (off > ws_size) return;  // workspace too small; nothing safe to do

  unsigned short* xb    = (unsigned short*)(ws + o_xb);
  unsigned short* outsb = (unsigned short*)(ws + o_outs);
  unsigned short* wihb  = (unsigned short*)(ws + o_wih);
  unsigned short* whhb  = (unsigned short*)(ws + o_whh);
  unsigned short* woutb = (unsigned short*)(ws + o_wout);
  float*          gx    = (float*)(ws + o_gx);
  float*          hfb[2]= { (float*)(ws + o_hf0), (float*)(ws + o_hf1) };
  unsigned short* hbb[2]= { (unsigned short*)(ws + o_hb0), (unsigned short*)(ws + o_hb1) };
  float*          lse   = (float*)(ws + o_lse);

  // 1) weight conversions f32 -> bf16
  cvt_f32_bf16<<<(G3H * HH + 255) / 256, 256, 0, stream>>>(w_ih, wihb, G3H * HH);
  cvt_f32_bf16<<<(G3H * HH + 255) / 256, 256, 0, stream>>>(w_hh, whhb, G3H * HH);
  cvt_f32_bf16<<<(VV * HH + 255) / 256, 256, 0, stream>>>(w_out, woutb, VV * HH);

  // 2) embed + relu + bf16 ; h0 init
  embed_relu_bf16<<<(MM * HH + 255) / 256, 256, 0, stream>>>(target, emb, xb);
  init_h<<<(BB * HH + 255) / 256, 256, 0, stream>>>(hidden, hfb[0], hbb[0]);

  // 3) gx = relu(emb) @ w_ih^T + b_ih   (WMMA bf16)
  gemm_nt_bf16<<<dim3(G3H / 128, MM / 128), 256, 0, stream>>>(
      xb, wihb, b_ih, gx, MM, G3H, HH);

  // 4) GRU scan, 10 dependent steps (ping-pong h buffers)
  for (int t = 0; t < TT; ++t) {
    gru_step<<<HH / 64, 256, 0, stream>>>(
        hbb[t & 1], hfb[t & 1], whhb, b_hh, gx,
        hfb[(t + 1) & 1], hbb[(t + 1) & 1], outsb, t);
  }

  // 5) logits = outs @ w_out^T + b_out  -> d_out (dominant WMMA GEMM)
  gemm_nt_bf16<<<dim3((VV + 127) / 128, MM / 128), 256, 0, stream>>>(
      outsb, woutb, b_out, out, MM, VV, HH);

  // 6) log_softmax: per-row lse then subtract
  row_lse<<<MM, 256, 0, stream>>>(out, lse, VV);
  sub_lse<<<dim3((VV + 255) / 256, MM), 256, 0, stream>>>(out, lse, VV);

  // 7) h_last (after t=9, parity lands in buffer 0)
  copy_hlast<<<(BB * HH + 255) / 256, 256, 0, stream>>>(hfb[0], out + (size_t)MM * VV);
}
